// PointNetSetAbstraction_3650722202242
// MI455X (gfx1250) — matrix operand order repaired and checked by reading, not compiled, not run on hardware
//
#include <hip/hip_runtime.h>

// ---------------- problem constants ----------------
#define BB   8
#define NN   8192
#define SS   1024
#define KK   32
#define CIN  9
#define M_ROWS (BB*SS*KK)        // 262144
#define RAD2 0.04f
#define EPSV 1e-5f

typedef __attribute__((ext_vector_type(16))) _Float16 v16h;
typedef __attribute__((ext_vector_type(8)))  _Float16 v8h;
typedef __attribute__((ext_vector_type(8)))  float    v8f;

// =====================================================================
// 1) Farthest point sampling: one block (1024 thr) per batch.
//    Each thread owns 8 points in registers; argmax via shfl + LDS tree.
// =====================================================================
__global__ __launch_bounds__(1024)
void fps_kernel(const float* __restrict__ xyz, int* __restrict__ fps_idx)
{
    const int b = blockIdx.x;
    const int t = threadIdx.x;
    const float* base = xyz + (long)b * NN * 3;

    float px[8], py[8], pz[8], dist[8];
#pragma unroll
    for (int i = 0; i < 8; ++i) {
        int n = i * 1024 + t;
        px[i] = base[n*3+0]; py[i] = base[n*3+1]; pz[i] = base[n*3+2];
        dist[i] = 1e10f;
    }

    __shared__ float swd[32];
    __shared__ int   swi[32];
    __shared__ int   s_far;

    int far = 0;
    for (int s = 0; s < SS; ++s) {
        if (t == 0) fps_idx[b*SS + s] = far;
        const float cx = base[far*3+0], cy = base[far*3+1], cz = base[far*3+2];

        float bd = -1.0f; int bi = 0;
#pragma unroll
        for (int i = 0; i < 8; ++i) {
            float dx = px[i]-cx, dy = py[i]-cy, dz = pz[i]-cz;
            float d  = dx*dx + dy*dy + dz*dz;
            dist[i]  = fminf(dist[i], d);
            int n = i * 1024 + t;
            if (dist[i] > bd) { bd = dist[i]; bi = n; }
        }
        // intra-wave argmax (ties -> lowest index, matching jnp.argmax)
#pragma unroll
        for (int off = 16; off > 0; off >>= 1) {
            float od = __shfl_xor(bd, off);
            int   oi = __shfl_xor(bi, off);
            if (od > bd || (od == bd && oi < bi)) { bd = od; bi = oi; }
        }
        if ((t & 31) == 0) { swd[t >> 5] = bd; swi[t >> 5] = bi; }
        __syncthreads();
        if (t < 32) {
            bd = swd[t]; bi = swi[t];
#pragma unroll
            for (int off = 16; off > 0; off >>= 1) {
                float od = __shfl_xor(bd, off);
                int   oi = __shfl_xor(bi, off);
                if (od > bd || (od == bd && oi < bi)) { bd = od; bi = oi; }
            }
            if (t == 0) s_far = bi;
        }
        __syncthreads();
        far = s_far;
    }
}

// =====================================================================
// 2) Gather centroid coordinates -> d_out[0 : B*S*3]
// =====================================================================
__global__ void gather_newxyz(const float* __restrict__ xyz,
                              const int* __restrict__ fps_idx,
                              float* __restrict__ out)
{
    int i = blockIdx.x * 256 + threadIdx.x;
    if (i >= BB * SS) return;
    int b = i / SS;
    int n = fps_idx[i];
    const float* p = xyz + ((long)b * NN + n) * 3;
    out[i*3+0] = p[0]; out[i*3+1] = p[1]; out[i*3+2] = p[2];
}

// =====================================================================
// 3) Ball query: one wave32 per query; ballot + prefix popcount keeps
//    the first K in-radius indices in ascending order; pad with first.
// =====================================================================
__global__ __launch_bounds__(256)
void ball_query(const float* __restrict__ xyz,
                const float* __restrict__ newxyz,
                int* __restrict__ gidx)
{
    const int lane = threadIdx.x & 31;
    const int q    = blockIdx.x * 8 + (threadIdx.x >> 5);   // [0, B*S)
    const int b    = q / SS;
    const float cx = newxyz[q*3+0], cy = newxyz[q*3+1], cz = newxyz[q*3+2];
    const float* base = xyz + (long)b * NN * 3;
    int* out = gidx + (long)q * KK;

    int cnt = 0, first = 0;
    for (int chunk = 0; chunk < NN/32; ++chunk) {
        int n = chunk * 32 + lane;
        if (chunk + 1 < NN/32)
            __builtin_prefetch(base + (n + 32) * 3, 0, 3);
        float dx = base[n*3+0]-cx, dy = base[n*3+1]-cy, dz = base[n*3+2]-cz;
        bool in = (dx*dx + dy*dy + dz*dz) <= RAD2;
        unsigned mask = (unsigned)__ballot(in);
        if (cnt == 0 && mask) first = chunk*32 + __ffs(mask) - 1;
        int pos = cnt + __popc(mask & ((1u << lane) - 1u));
        if (in && pos < KK) out[pos] = n;
        cnt += __popc(mask);
        if (cnt >= KK) break;
    }
    if (lane < KK && lane >= cnt) out[lane] = first;
}

// =====================================================================
// 4) Build MLP input rows: [points(6), xyz - centroid(3), zero pad] as
//    f16, padded K-dim to 32 for the 16x16x32 WMMA.
// =====================================================================
__global__ void group_build(const float* __restrict__ xyz,
                            const float* __restrict__ pts,
                            const int*  __restrict__ gidx,
                            const float* __restrict__ newxyz,
                            _Float16* __restrict__ X0)
{
    int m = blockIdx.x * 256 + threadIdx.x;
    if (m >= M_ROWS) return;
    int bs = m / KK;
    int b  = bs / SS;
    int n  = gidx[m];
    const float* p = pts    + ((long)b * NN + n) * 6;
    const float* x = xyz    + ((long)b * NN + n) * 3;
    const float* c = newxyz + (long)bs * 3;
    _Float16* row = X0 + (long)m * 32;
#pragma unroll
    for (int j = 0; j < 6; ++j) row[j] = (_Float16)p[j];
    row[6] = (_Float16)(x[0] - c[0]);
    row[7] = (_Float16)(x[1] - c[1]);
    row[8] = (_Float16)(x[2] - c[2]);
#pragma unroll
    for (int j = 9; j < 32; ++j) row[j] = (_Float16)0.0f;
}

// =====================================================================
// 5) Fused WMMA GEMM:
//      A-load  : optional BN-normalize+ReLU of previous layer (per-channel
//                scale/shift staged in LDS), applied to each f16 chunk.
//      compute : v_wmma_f32_16x16x32_f16, 8 waves x 16-row stripes.
//      epilogue: per-channel sum / sumsq partials via LDS ds_add_f32,
//                flushed with one global atomicAdd per channel per block.
//    Y is stored as raw (pre-norm) f16; the *next* consumer normalizes.
// =====================================================================
__global__ void zero_stats(float* stats) { stats[threadIdx.x] = 0.0f; }

template<int CK, int CN, bool NORM_IN>
__global__ __launch_bounds__(256)
void gemm_wmma(const _Float16* __restrict__ X,
               const float* __restrict__ W, int creal,
               const float* __restrict__ bias,
               const float* __restrict__ statsIn,
               const float* __restrict__ gamIn,
               const float* __restrict__ betIn,
               _Float16* __restrict__ Y,
               float* __restrict__ statsOut)
{
    __shared__ __align__(16) _Float16 wlds[CN * CK];   // W transposed: [n][k]
    __shared__ float sclds[CK], shlds[CK];             // input-norm scale/shift
    __shared__ float r1lds[CN], r2lds[CN];             // per-channel partials

    for (int i = threadIdx.x; i < CN * CK; i += 256) {
        int n = i / CK, k = i % CK;
        wlds[i] = (k < creal) ? (_Float16)W[k * CN + n] : (_Float16)0.0f;
    }
    if constexpr (NORM_IN) {
        const float invM = 1.0f / (float)M_ROWS;
        for (int c = threadIdx.x; c < CK; c += 256) {
            float mean = statsIn[c] * invM;
            float var  = statsIn[128 + c] * invM - mean * mean;
            float s    = gamIn[c] * rsqrtf(var + EPSV);
            sclds[c] = s;
            shlds[c] = betIn[c] - mean * s;
        }
    }
    for (int c = threadIdx.x; c < CN; c += 256) { r1lds[c] = 0.0f; r2lds[c] = 0.0f; }
    __syncthreads();

    const int wave = threadIdx.x >> 5;
    const int lane = threadIdx.x & 31;
    const int hid  = lane >> 4;     // lane half
    const int lr   = lane & 15;
    const long rowBase = (long)blockIdx.x * 128 + wave * 16;

    constexpr int NT = CN / 16;
    v8f acc[NT];
#pragma unroll
    for (int t = 0; t < NT; ++t) {
        float bv = bias[t * 16 + lr];
#pragma unroll
        for (int j = 0; j < 8; ++j) acc[t][j] = bv;
    }

#pragma unroll
    for (int kk = 0; kk < CK; kk += 32) {
        // A fragment (16x32 f16): half 0 -> K 0..7 & 16..23, half 1 -> 8..15 & 24..31
        const v8h* ap = (const v8h*)(X + (rowBase + lr) * CK + kk);
        v8h a0 = ap[hid];
        v8h a1 = ap[2 + hid];
        if constexpr (NORM_IN) {
#pragma unroll
            for (int e = 0; e < 8; ++e) {
                int c0 = kk + hid * 8 + e;
                int c1 = kk + 16 + hid * 8 + e;
                a0[e] = (_Float16)fmaxf((float)a0[e] * sclds[c0] + shlds[c0], 0.0f);
                a1[e] = (_Float16)fmaxf((float)a1[e] * sclds[c1] + shlds[c1], 0.0f);
            }
        }
        v16h a = __builtin_shufflevector(a0, a1, 0,1,2,3,4,5,6,7,8,9,10,11,12,13,14,15);
#pragma unroll
        for (int t = 0; t < NT; ++t) {
            // B fragment (32x16 f16): col = lane&15, half 0 -> K 0..7 & 16..23, half 1 -> 8..15 & 24..31
            const _Float16* wp = &wlds[(t * 16 + lr) * CK + kk + hid * 8];
            v8h b0 = *(const v8h*)(wp);
            v8h b1 = *(const v8h*)(wp + 16);
            v16h bm = __builtin_shufflevector(b0, b1, 0,1,2,3,4,5,6,7,8,9,10,11,12,13,14,15);
            acc[t] = __builtin_amdgcn_wmma_f32_16x16x32_f16(
                false, a, false, bm, (short)0, acc[t], false, false);
        }
    }

    // store Y (raw, pre-norm) + accumulate per-channel stats
#pragma unroll
    for (int t = 0; t < NT; ++t) {
        long row = rowBase + hid * 8;
        int  col = t * 16 + lr;
        float p1 = 0.0f, p2 = 0.0f;
#pragma unroll
        for (int j = 0; j < 8; ++j) {
            float v = acc[t][j];
            p1 += v; p2 += v * v;
            Y[(row + j) * CN + col] = (_Float16)v;
        }
        atomicAdd(&r1lds[col], p1);
        atomicAdd(&r2lds[col], p2);
    }
    __syncthreads();
    for (int c = threadIdx.x; c < CN; c += 256) {
        atomicAdd(&statsOut[c],       r1lds[c]);
        atomicAdd(&statsOut[128 + c], r2lds[c]);
    }
}

// =====================================================================
// 6) Fused final stage: BN-normalize layer-2 output on read, ReLU,
//    max over the K neighbor axis, write f32 output.
// =====================================================================
__global__ __launch_bounds__(256)
void max_norm_k(const _Float16* __restrict__ Y, const float* __restrict__ stats,
                const float* __restrict__ gam, const float* __restrict__ bet,
                float* __restrict__ out)
{
    int i = blockIdx.x * 256 + threadIdx.x;   // B*S*128
    int bs = i >> 7, c = i & 127;
    const float invM = 1.0f / (float)M_ROWS;
    float mean = stats[c] * invM;
    float var  = stats[128 + c] * invM - mean * mean;
    float sc   = gam[c] * rsqrtf(var + EPSV);
    float sh   = bet[c] - mean * sc;
    const _Float16* p = Y + (long)bs * KK * 128 + c;
    float m = 0.0f;   // post-ReLU values are >= 0
#pragma unroll
    for (int k = 0; k < KK; ++k)
        m = fmaxf(m, (float)p[k * 128] * sc + sh);
    out[i] = m;
}

// =====================================================================
// host-side orchestration
// =====================================================================
extern "C" void kernel_launch(void* const* d_in, const int* in_sizes, int n_in,
                              void* d_out, int out_size, void* d_ws, size_t ws_size,
                              hipStream_t stream) {
    const float* xyz = (const float*)d_in[0];
    const float* pts = (const float*)d_in[1];
    const float* W0  = (const float*)d_in[2];
    const float* b0  = (const float*)d_in[3];
    const float* g0  = (const float*)d_in[4];
    const float* be0 = (const float*)d_in[5];
    const float* W1  = (const float*)d_in[6];
    const float* b1  = (const float*)d_in[7];
    const float* g1  = (const float*)d_in[8];
    const float* be1 = (const float*)d_in[9];
    const float* W2  = (const float*)d_in[10];
    const float* b2  = (const float*)d_in[11];
    const float* g2  = (const float*)d_in[12];
    const float* be2 = (const float*)d_in[13];

    float* out_xyz = (float*)d_out;                 // (B,S,3)
    float* out_pts = out_xyz + (long)BB * SS * 3;   // (B,S,128)

    // workspace layout (all offsets 256B aligned); total ~103 MB < 192 MB L2
    char* ws = (char*)d_ws;
    int*      fpsIdx = (int*)ws;                                       //   32 KB
    int*      gidx   = (int*)(ws + 32768);                             //    1 MB
    _Float16* bufA   = (_Float16*)(ws + 32768 + 1048576);              // 33.5 MB  (<= M x 64)
    _Float16* bufB   = (_Float16*)(ws + 32768 + 1048576 + 33554432);   //   67 MB  (<= M x 128)
    float*    stats  = (float*)(ws + 32768 + 1048576 + 33554432 + 67108864); // 3 x 256 floats
    float* statsA = stats;        // layer-0 output stats
    float* statsB = stats + 256;  // layer-1 output stats
    float* statsC = stats + 512;  // layer-2 output stats

    // --- sampling / grouping ---
    fps_kernel   <<<BB, 1024, 0, stream>>>(xyz, fpsIdx);
    gather_newxyz<<<(BB*SS + 255) / 256, 256, 0, stream>>>(xyz, fpsIdx, out_xyz);
    ball_query   <<<BB*SS / 8, 256, 0, stream>>>(xyz, out_xyz, gidx);
    group_build  <<<M_ROWS / 256, 256, 0, stream>>>(xyz, pts, gidx, out_xyz, bufA);

    // --- layer 0: 9 (pad 32) -> 64 : X0(bufA) -> Y0(bufB), stats -> statsA
    zero_stats<<<1, 256, 0, stream>>>(statsA);
    gemm_wmma<32, 64, false><<<M_ROWS / 128, 256, 0, stream>>>(
        bufA, W0, CIN, b0, nullptr, nullptr, nullptr, bufB, statsA);

    // --- layer 1: 64 -> 64 : norm(Y0) on load -> Y1(bufA), stats -> statsB
    zero_stats<<<1, 256, 0, stream>>>(statsB);
    gemm_wmma<64, 64, true><<<M_ROWS / 128, 256, 0, stream>>>(
        bufB, W1, 64, b1, statsA, g0, be0, bufA, statsB);

    // --- layer 2: 64 -> 128 : norm(Y1) on load -> Y2(bufB), stats -> statsC
    zero_stats<<<1, 256, 0, stream>>>(statsC);
    gemm_wmma<64, 128, true><<<M_ROWS / 128, 256, 0, stream>>>(
        bufA, W2, 64, b2, statsB, g1, be1, bufB, statsC);

    // --- fused normalize + ReLU + max over neighbors ---
    max_norm_k<<<BB*SS*128 / 256, 256, 0, stream>>>(bufB, statsC, g2, be2, out_pts);
}
